// Model_45741401703050
// MI455X (gfx1250) — matrix operand order repaired and checked by reading, not compiled, not run on hardware
//
#include <hip/hip_runtime.h>
#include <hip/hip_bf16.h>

// ---------------------------------------------------------------------------
// Hetero GraphSAGE (6 layers, H=128) for MI455X / gfx1250.
//
// Roofline: ~0.57 TFLOP of GEMM vs ~15-20 GB HBM traffic -> memory bound
// (~0.8 ms at 23.3 TB/s).  GEMMs: v_wmma_f32_16x16x32_bf16, fp32 accumulate,
// native v_cvt_pk_bf16_f32 operand conversion.  Each wave holds one 16-row A
// fragment and sweeps all N-tiles (8x less A traffic); W is staged once per
// block into LDS as bf16.  B fragments are consumed in groups of 4 to cap
// live registers (~32 for B) so the 4 A-chunk global_load_b128s can issue
// concurrently into distinct registers instead of being serialized through a
// recycled quad.  K is a template parameter (128/256) -> fully unrolled.
//
// d_in layout (setup_inputs() insertion order, then params flattened as a
// JAX pytree: dict keys sorted alphabetically):
//   0 cases_node_id 1 app_node_id 2 def_node_id 3 app_x 4 def_x
//   5..8 ei_app_cases, ei_def_cases, ei_cases_app, ei_cases_def [2,500000]
//   9 app_emb 10 app_lin_b 11 app_lin_w 12 cases_emb 13 def_emb
//   14 def_lin_b 15 def_lin_w
//   16..87 layers[6] x {applicants__cases, cases__applicants,
//                       cases__defendants, defendants__cases} x {bl, wl, wr}
//   88..93 out: applicants{b,w}, cases{b,w}, defendants{b,w}
// ---------------------------------------------------------------------------

typedef __attribute__((ext_vector_type(16))) __bf16 v16bf;
typedef __attribute__((ext_vector_type(8)))  float  v8f;

#define HDIM 128

// Lowered by clang to the native bf16 convert (v_cvt_pk_bf16_f32).
static __device__ inline __bf16 f2bf(float f) { return (__bf16)f; }

// ---------------------------------------------------------------------------
// Out[M, Nout] (+)= (A * rowscale) @ W^T + bias + addRows[addIdx]
//   A: [M,K] f32 row-major; W: [Nout,K] f32 row-major; NT = ceil(Nout/16).
// Block = 128 threads (4 waves); wave w owns rows (blockIdx.x*4+w)*16..+15.
// Dynamic LDS: NT*16*K bf16 = staged, zero-padded W tile.
// ---------------------------------------------------------------------------
template <int NT, int K>
__global__ __launch_bounds__(128)
void wmma_gemm(const float* __restrict__ A,
               const float* __restrict__ rownorm,
               const float* __restrict__ W,
               const float* __restrict__ bias,
               const float* __restrict__ addRows,
               const int*   __restrict__ addIdx,
               float* __restrict__ Out,
               int M, int Nout, int accumulate) {
  extern __shared__ __bf16 sW[];                 // [NT*16, K] bf16
  const int tid  = threadIdx.x;
  const int lane = tid & 31;
  const int wave = tid >> 5;
  const int half = lane >> 4;                    // K-half selector
  const int l15  = lane & 15;

  // ---- stage W tile (bf16, OOB rows zeroed) into LDS ----
  for (int idx = tid * 4; idx < NT * 16 * K; idx += 128 * 4) {
    int n = idx / K, k = idx - n * K;            // K % 4 == 0
    int nc = n < Nout ? n : Nout - 1;
    float msk = n < Nout ? 1.0f : 0.0f;
    float4 w = *(const float4*)(W + (size_t)nc * K + k);
    __bf16* p = sW + (size_t)n * K + k;
    p[0] = f2bf(w.x * msk); p[1] = f2bf(w.y * msk);
    p[2] = f2bf(w.z * msk); p[3] = f2bf(w.w * msk);
  }
  __syncthreads();

  const int tileM = (blockIdx.x * 4 + wave) * 16;
  if (tileM >= M) return;                        // wave-uniform, after barrier

  const int mA = tileM + l15;                    // A row held by this lane
  const int mC = mA < M ? mA : M - 1;            // clamped (always valid)
  float maskA = (mA < M) ? 1.0f : 0.0f;
  if (rownorm) maskA *= 1.0f / fmaxf(rownorm[mC], 1.0f);   // segment-mean
  const float* __restrict__ arow = A + (size_t)mC * K;
  const __bf16* __restrict__ brow = sW + (size_t)l15 * K + half * 16;

  v8f zero = {};
  v8f acc[NT];
#pragma unroll
  for (int nt = 0; nt < NT; ++nt) acc[nt] = zero;

#pragma unroll
  for (int k0 = 0; k0 < K; k0 += 32) {
    // A 16x32 bf16 fragment: one contiguous 16-float chunk per lane:
    // [k0+half*16, +16) -> 4x concurrent
    // global_load_b128 into distinct registers.
    float4 a0 = *(const float4*)(arow + k0 + half * 16);
    float4 a1 = *(const float4*)(arow + k0 + half * 16 + 4);
    float4 a2 = *(const float4*)(arow + k0 + half * 16 + 8);
    float4 a3 = *(const float4*)(arow + k0 + half * 16 + 12);
    v16bf a;
    a[0]  = f2bf(a0.x * maskA); a[1]  = f2bf(a0.y * maskA);
    a[2]  = f2bf(a0.z * maskA); a[3]  = f2bf(a0.w * maskA);
    a[4]  = f2bf(a1.x * maskA); a[5]  = f2bf(a1.y * maskA);
    a[6]  = f2bf(a1.z * maskA); a[7]  = f2bf(a1.w * maskA);
    a[8]  = f2bf(a2.x * maskA); a[9]  = f2bf(a2.y * maskA);
    a[10] = f2bf(a2.z * maskA); a[11] = f2bf(a2.w * maskA);
    a[12] = f2bf(a3.x * maskA); a[13] = f2bf(a3.y * maskA);
    a[14] = f2bf(a3.z * maskA); a[15] = f2bf(a3.w * maskA);

    // Consume B tiles in groups of 4: caps live B registers at 32 while
    // still letting the ds_loads of a group overlap the WMMA chain.
#pragma unroll
    for (int g = 0; g < NT; g += 4) {
      v16bf bg[4];
#pragma unroll
      for (int j = 0; j < 4; ++j) {
        if (g + j < NT)
          bg[j] = *(const v16bf*)(brow + (size_t)(g + j) * 16 * K + k0);
      }
#pragma unroll
      for (int j = 0; j < 4; ++j) {
        if (g + j < NT)
          acc[g + j] = __builtin_amdgcn_wmma_f32_16x16x32_bf16(
              false, a, false, bg[j], (short)0, acc[g + j], false, false);
      }
    }
  }

  // ---- epilogue: C/D layout lanes 0-15 rows 0..7, lanes 16-31 rows 8..15 ---
#pragma unroll
  for (int nt = 0; nt < NT; ++nt) {
    int nB = nt * 16 + l15;
    if (nB >= Nout) continue;
    float bv = bias ? bias[nB] : 0.0f;
#pragma unroll
    for (int i = 0; i < 8; ++i) {
      int m = tileM + (half ? 8 : 0) + i;
      if (m >= M) continue;
      size_t o = (size_t)m * Nout + nB;
      float v = acc[nt][i] + bv;
      if (addRows) {
        int r = addIdx ? addIdx[m] : m;
        v += addRows[(size_t)r * Nout + nB];
      }
      if (accumulate) v += Out[o];
      Out[o] = v;
    }
  }
}

// -------------------- segment-sum (mean numerator + counts) ----------------
__global__ void edge_count(const int* __restrict__ dst, int E,
                           float* __restrict__ cnt) {
  int e = blockIdx.x * blockDim.x + threadIdx.x;
  if (e < E) unsafeAtomicAdd(&cnt[dst[e]], 1.0f);
}

__global__ void edge_scatter(const int* __restrict__ src,
                             const int* __restrict__ dst,
                             const float* __restrict__ X,
                             float* __restrict__ agg, int E) {
  long long t = (long long)blockIdx.x * blockDim.x + threadIdx.x;
  int e = (int)(t >> 7);
  int h = (int)(t & (HDIM - 1));
  if (e < E) {
    unsafeAtomicAdd(&agg[(size_t)dst[e] * HDIM + h],
                    X[(size_t)src[e] * HDIM + h]);
  }
}

// -------------------- small elementwise helpers ----------------------------
__global__ void relu_copy(const float* __restrict__ in, float* __restrict__ out,
                          long long n) {
  long long i = (long long)blockIdx.x * blockDim.x + threadIdx.x;
  if (i < n) out[i] = fmaxf(in[i], 0.0f);
}

__global__ void gather_rows128(const float* __restrict__ tab,
                               const int* __restrict__ idx,
                               float* __restrict__ out, int M) {
  long long t = (long long)blockIdx.x * blockDim.x + threadIdx.x;
  int m = (int)(t >> 7);
  int h = (int)(t & (HDIM - 1));
  if (m < M) out[(size_t)m * HDIM + h] = tab[(size_t)idx[m] * HDIM + h];
}

// ---------------------------------------------------------------------------
extern "C" void kernel_launch(void* const* d_in, const int* in_sizes, int n_in,
                              void* d_out, int out_size, void* d_ws,
                              size_t ws_size, hipStream_t stream) {
  (void)in_sizes; (void)n_in; (void)out_size; (void)ws_size;
  const int NC = 200000, NA = 150000, ND = 150000, NE = 500000;
  const int EMB = 256, NCLS = 10, L = 6;

  // ---- workspace carve-up (~615 MB) ----
  char* ws = (char*)d_ws;
  size_t off = 0;
  auto carve = [&](size_t bytes) -> void* {
    void* p = ws + off;
    off += (bytes + 255) & ~(size_t)255;
    return p;
  };
  float* xC  = (float*)carve((size_t)NC * HDIM * 4);
  float* xA  = (float*)carve((size_t)NA * HDIM * 4);
  float* xD  = (float*)carve((size_t)ND * HDIM * 4);
  float* nC  = (float*)carve((size_t)NC * HDIM * 4);
  float* nA  = (float*)carve((size_t)NA * HDIM * 4);
  float* nD  = (float*)carve((size_t)ND * HDIM * 4);
  float* agg = (float*)carve((size_t)NC * HDIM * 4);   // max dest = cases
  float* cnt = (float*)carve((size_t)NC * 4);

  // ---- inputs ----
  const int*   cases_id  = (const int*)d_in[0];
  const int*   app_id    = (const int*)d_in[1];
  const int*   def_id    = (const int*)d_in[2];
  const float* app_x     = (const float*)d_in[3];
  const float* def_x     = (const float*)d_in[4];
  const int*   ei[4]     = {(const int*)d_in[5], (const int*)d_in[6],
                            (const int*)d_in[7], (const int*)d_in[8]};
  const float* app_emb   = (const float*)d_in[9];
  const float* app_lin_b = (const float*)d_in[10];
  const float* app_lin_w = (const float*)d_in[11];
  const float* cases_emb = (const float*)d_in[12];
  const float* def_emb   = (const float*)d_in[13];
  const float* def_lin_b = (const float*)d_in[14];
  const float* def_lin_w = (const float*)d_in[15];

  dim3 blk(128);
  auto gGrid = [](int M) { return dim3((unsigned)((M + 63) / 64)); };
  const size_t shProj = (size_t)8 * 16 * EMB  * sizeof(__bf16);   // 64 KB
  const size_t shH    = (size_t)8 * 16 * HDIM * sizeof(__bf16);   // 32 KB
  const size_t shOut  = (size_t)1 * 16 * HDIM * sizeof(__bf16);   //  4 KB

  // ---- input projections ----
  gather_rows128<<<((long long)NC * HDIM + 255) / 256, 256, 0, stream>>>(
      cases_emb, cases_id, xC, NC);
  wmma_gemm<8, 256><<<gGrid(NA), blk, shProj, stream>>>(
      app_x, nullptr, app_lin_w, app_lin_b, app_emb, app_id, xA,
      NA, HDIM, 0);
  wmma_gemm<8, 256><<<gGrid(ND), blk, shProj, stream>>>(
      def_x, nullptr, def_lin_w, def_lin_b, def_emb, def_id, xD,
      ND, HDIM, 0);

  // ---- per-edge-type tables (order matches ei[]) ----
  const int pgroup[4] = {0, 3, 1, 2};  // sorted param-dict group index
  float* xSrc[4];
  float* xDst[4];
  float* newDst[4];
  int    Nd[4] = {NC, NC, NA, ND};

  for (int li = 0; li < L; ++li) {
    xSrc[0] = xA; xSrc[1] = xD; xSrc[2] = xC; xSrc[3] = xC;
    xDst[0] = xC; xDst[1] = xC; xDst[2] = xA; xDst[3] = xD;
    newDst[0] = nC; newDst[1] = nC; newDst[2] = nA; newDst[3] = nD;

    hipMemsetAsync(nC, 0, (size_t)NC * HDIM * 4, stream);
    hipMemsetAsync(nA, 0, (size_t)NA * HDIM * 4, stream);
    hipMemsetAsync(nD, 0, (size_t)ND * HDIM * 4, stream);

    for (int t = 0; t < 4; ++t) {
      const int base = 16 + (li * 4 + pgroup[t]) * 3;  // {bl, wl, wr}
      const float* bl = (const float*)d_in[base + 0];
      const float* wl = (const float*)d_in[base + 1];
      const float* wr = (const float*)d_in[base + 2];
      const int* eSrc = ei[t];
      const int* eDst = ei[t] + NE;

      hipMemsetAsync(agg, 0, (size_t)Nd[t] * HDIM * 4, stream);
      hipMemsetAsync(cnt, 0, (size_t)Nd[t] * 4, stream);
      edge_count<<<(NE + 255) / 256, 256, 0, stream>>>(eDst, NE, cnt);
      edge_scatter<<<((long long)NE * HDIM + 255) / 256, 256, 0, stream>>>(
          eSrc, eDst, xSrc[t], agg, NE);

      // new[d] += (agg/cnt) @ wl.T + bl
      wmma_gemm<8, HDIM><<<gGrid(Nd[t]), blk, shH, stream>>>(
          agg, cnt, wl, bl, nullptr, nullptr, newDst[t],
          Nd[t], HDIM, 1);
      // new[d] += x[d] @ wr.T
      wmma_gemm<8, HDIM><<<gGrid(Nd[t]), blk, shH, stream>>>(
          xDst[t], nullptr, wr, nullptr, nullptr, nullptr, newDst[t],
          Nd[t], HDIM, 1);
    }

    relu_copy<<<((long long)NC * HDIM + 255) / 256, 256, 0, stream>>>(
        nC, xC, (long long)NC * HDIM);
    relu_copy<<<((long long)NA * HDIM + 255) / 256, 256, 0, stream>>>(
        nA, xA, (long long)NA * HDIM);
    relu_copy<<<((long long)ND * HDIM + 255) / 256, 256, 0, stream>>>(
        nD, xD, (long long)ND * HDIM);
  }

  // ---- output heads (sorted: applicants{b,w}, cases{b,w}, defendants{b,w})
  const float* outb_app = (const float*)d_in[88];
  const float* outw_app = (const float*)d_in[89];
  const float* outb_cas = (const float*)d_in[90];
  const float* outw_cas = (const float*)d_in[91];
  const float* outb_def = (const float*)d_in[92];
  const float* outw_def = (const float*)d_in[93];
  float* out = (float*)d_out;

  wmma_gemm<1, HDIM><<<gGrid(NC), blk, shOut, stream>>>(
      xC, nullptr, outw_cas, outb_cas, nullptr, nullptr, out,
      NC, NCLS, 0);
  wmma_gemm<1, HDIM><<<gGrid(NA), blk, shOut, stream>>>(
      xA, nullptr, outw_app, outb_app, nullptr, nullptr,
      out + (size_t)NC * NCLS, NA, NCLS, 0);
  wmma_gemm<1, HDIM><<<gGrid(ND), blk, shOut, stream>>>(
      xD, nullptr, outw_def, outb_def, nullptr, nullptr,
      out + (size_t)(NC + NA) * NCLS, ND, NCLS, 0);
}
